// GPT_51900384805412
// MI455X (gfx1250) — matrix operand order repaired and checked by reading, not compiled, not run on hardware
//
#include <hip/hip_runtime.h>

typedef __attribute__((ext_vector_type(16))) _Float16 v16h;
typedef __attribute__((ext_vector_type(8)))  _Float16 v8h;
typedef __attribute__((ext_vector_type(8)))  float    v8f;

union HF16 { v16h v; v8h h[2]; };

#define WMMA_F16(a, b, c) \
  __builtin_amdgcn_wmma_f32_16x16x32_f16(false, (a), false, (b), (short)0, (c), false, false)

constexpr int E_ = 768;
constexpr int S_ = 1024;
constexpr int H_ = 12;
constexpr int D_ = 64;
constexpr int L_ = 12;
constexpr int V_ = 50257;
constexpr int NTOK = 2048;            // B * S

// ---------------------------------------------------------------------------
// Embedding: x[row] = tok_emb[tokens[row]] + pos_emb[row % S]
// ---------------------------------------------------------------------------
__global__ __launch_bounds__(256) void embed_kernel(
    const int* __restrict__ tokens, const float* __restrict__ tok_emb,
    const float* __restrict__ pos_emb, float* __restrict__ x)
{
  const int row = blockIdx.x;
  const int t   = threadIdx.x;
  const int tok = tokens[row];
  const int pos = row & (S_ - 1);
#pragma unroll
  for (int i = 0; i < 3; ++i) {
    const int c = t + i * 256;
    x[(size_t)row * E_ + c] =
        tok_emb[(size_t)tok * E_ + c] + pos_emb[(size_t)pos * E_ + c];
  }
}

// ---------------------------------------------------------------------------
// LayerNorm (unbiased variance, ddof=1) -> f16 output
// ---------------------------------------------------------------------------
__global__ __launch_bounds__(256) void ln_kernel(
    const float* __restrict__ x, const float* __restrict__ sc,
    const float* __restrict__ bs, _Float16* __restrict__ y)
{
  __shared__ float red[18];
  const int row = blockIdx.x;
  const int t = threadIdx.x;
  const float* xp = x + (size_t)row * E_;
  const float v0 = xp[t], v1 = xp[t + 256], v2 = xp[t + 512];
  float s = v0 + v1 + v2;
  float q = v0 * v0 + v1 * v1 + v2 * v2;
#pragma unroll
  for (int m = 1; m < 32; m <<= 1) {
    s += __shfl_xor(s, m, 32);
    q += __shfl_xor(q, m, 32);
  }
  const int wave = t >> 5, lane = t & 31;
  if (lane == 0) { red[wave] = s; red[8 + wave] = q; }
  __syncthreads();
  if (t == 0) {
    float S = 0.f, Q = 0.f;
#pragma unroll
    for (int w = 0; w < 8; ++w) { S += red[w]; Q += red[8 + w]; }
    const float mean = S * (1.f / 768.f);
    const float var  = (Q - 768.f * mean * mean) * (1.f / 767.f);
    red[16] = mean;
    red[17] = rsqrtf(var + 1e-5f);
  }
  __syncthreads();
  const float mean = red[16], rstd = red[17];
  y[(size_t)row * E_ + t]       = (_Float16)(sc[t]       * (v0 - mean) * rstd + bs[t]);
  y[(size_t)row * E_ + t + 256] = (_Float16)(sc[t + 256] * (v1 - mean) * rstd + bs[t + 256]);
  y[(size_t)row * E_ + t + 512] = (_Float16)(sc[t + 512] * (v2 - mean) * rstd + bs[t + 512]);
}

// ---------------------------------------------------------------------------
// V transpose: v16[b*S+s][h*64+d] -> vt[(b*H+h)*64+d][s]   (64x64 LDS tiles)
// ---------------------------------------------------------------------------
__global__ __launch_bounds__(256) void vtrans_kernel(
    const _Float16* __restrict__ v, _Float16* __restrict__ vt)
{
  __shared__ _Float16 tile[64][72];
  const int t  = threadIdx.x;
  const int s0 = blockIdx.x * 64;
  const int h  = blockIdx.y;            // E/64 == H, so col tile == head
  const int b  = blockIdx.z;
#pragma unroll
  for (int i = 0; i < 2; ++i) {
    const int e  = t + i * 256;         // 0..511 v8h chunks
    const int sr = e >> 3;              // 0..63
    const int co = (e & 7) * 8;         // 0..56
    const v8h val = *(const v8h*)(v + (size_t)(b * S_ + s0 + sr) * E_ + h * 64 + co);
    *(v8h*)(&tile[sr][co]) = val;
  }
  __syncthreads();
#pragma unroll
  for (int i = 0; i < 2; ++i) {
    const int e  = t + i * 256;
    const int dr = e >> 3;              // 0..63
    const int so = (e & 7) * 8;         // 0..56
    v8h val;
#pragma unroll
    for (int jj = 0; jj < 8; ++jj) val[jj] = tile[so + jj][dr];
    *(v8h*)(vt + ((size_t)(b * H_ + h) * 64 + dr) * S_ + s0 + so) = val;
  }
}

// ---------------------------------------------------------------------------
// WMMA GEMM: C[M,N] = A[M,K](f16) * Bw[K,N](fp32->f16) (+bias)(+GELU)(+res)
// Block tile 128x128, 8 waves (4 in M, 2 in N), K-step 32.
// ---------------------------------------------------------------------------
constexpr int BM = 128, BN = 128, BK = 32;
constexpr int ASTR = 40;   // halves per LDS row of A (16B aligned fragments)
constexpr int BSTR = 40;   // halves per LDS col-row of B (stored N-major)

template <bool GELU_, bool RES_, bool F16OUT_, bool NV4_>
__global__ __launch_bounds__(256) void gemm_kernel(
    const _Float16* __restrict__ A, const float* __restrict__ Bw,
    const float* __restrict__ bias, const float* __restrict__ resIn,
    float* __restrict__ outF, _Float16* __restrict__ outH,
    int M, int N, int K)
{
  __shared__ _Float16 As[BM * ASTR];
  __shared__ _Float16 Bs[BN * BSTR];

  const int t    = threadIdx.x;
  const int lane = t & 31, wave = t >> 5;
  const int nl   = lane & 15, hi = lane >> 4;
  const int wm   = wave >> 1;          // 0..3
  const int wn   = wave & 1;           // 0..1
  const int m0   = blockIdx.y * BM;
  const int n0   = blockIdx.x * BN;

  const v8f zero8 = {0.f, 0.f, 0.f, 0.f, 0.f, 0.f, 0.f, 0.f};
  v8f acc[2][4];
#pragma unroll
  for (int i = 0; i < 2; ++i)
#pragma unroll
    for (int j = 0; j < 4; ++j) acc[i][j] = zero8;

  for (int k0 = 0; k0 < K; k0 += BK) {
    __syncthreads();
    // ---- stage A: 128x32 halves, vector 16B chunks ----
#pragma unroll
    for (int i = 0; i < 2; ++i) {
      const int c = t + i * 256;             // 0..511
      const int row = c >> 2;
      const int ko  = (c & 3) * 8;
      const v8h av = *(const v8h*)(A + (size_t)(m0 + row) * K + k0 + ko);
      *(v8h*)(As + row * ASTR + ko) = av;
    }
    // ---- stage B: 32x128 fp32 -> f16, transposed to N-major ----
    if (NV4_) {
#pragma unroll
      for (int i = 0; i < 4; ++i) {
        const int c  = t + i * 256;          // 0..1023 float4 chunks
        const int kk = c >> 5;               // 0..31
        const int no = (c & 31) * 4;         // 0..124
        const float4 b4 = *(const float4*)(Bw + (size_t)(k0 + kk) * N + n0 + no);
        Bs[(no + 0) * BSTR + kk] = (_Float16)b4.x;
        Bs[(no + 1) * BSTR + kk] = (_Float16)b4.y;
        Bs[(no + 2) * BSTR + kk] = (_Float16)b4.z;
        Bs[(no + 3) * BSTR + kk] = (_Float16)b4.w;
      }
    } else {
#pragma unroll
      for (int i = 0; i < 16; ++i) {
        const int e  = t + i * 256;          // 0..4095
        const int kk = e >> 7;               // 0..31
        const int nn = e & 127;              // 0..127
        const int col = n0 + nn;
        const float bv = (col < N) ? Bw[(size_t)(k0 + kk) * N + col] : 0.f;
        Bs[nn * BSTR + kk] = (_Float16)bv;
      }
    }
    if (k0 + BK < K)
      __builtin_prefetch(Bw + (size_t)(k0 + BK) * N + n0 + (t & 127), 0, 1);
    __syncthreads();

    // ---- fragments from LDS ----
    HF16 af[2], bf[4];
#pragma unroll
    for (int i = 0; i < 2; ++i) {
      const _Float16* rp = As + (wm * 32 + i * 16 + nl) * ASTR;
      af[i].h[0] = *(const v8h*)(rp + hi * 8);
      af[i].h[1] = *(const v8h*)(rp + 16 + hi * 8);
    }
#pragma unroll
    for (int j = 0; j < 4; ++j) {
      const _Float16* cp = Bs + (wn * 64 + j * 16 + nl) * BSTR;
      bf[j].h[0] = *(const v8h*)(cp + hi * 8);
      bf[j].h[1] = *(const v8h*)(cp + 16 + hi * 8);
    }
#pragma unroll
    for (int i = 0; i < 2; ++i)
#pragma unroll
      for (int j = 0; j < 4; ++j)
        acc[i][j] = WMMA_F16(af[i].v, bf[j].v, acc[i][j]);
  }

  // ---- epilogue ----
#pragma unroll
  for (int i = 0; i < 2; ++i) {
#pragma unroll
    for (int j = 0; j < 4; ++j) {
#pragma unroll
      for (int r = 0; r < 8; ++r) {
        const int row = m0 + wm * 32 + i * 16 + r + 8 * hi;
        const int col = n0 + wn * 64 + j * 16 + nl;
        if (col < N) {
          float vv = acc[i][j][r];
          if (bias) vv += bias[col];
          if (GELU_) {
            const float u = vv;
            vv = 0.5f * u *
                 (1.f + tanhf(0.7978845608028654f * (u + 0.044715f * u * u * u)));
          }
          const size_t idx = (size_t)row * N + col;
          if (RES_)        outF[idx] = resIn[idx] + vv;
          else if (F16OUT_) outH[idx] = (_Float16)vv;
          else              outF[idx] = vv;
        }
      }
    }
  }
}

// ---------------------------------------------------------------------------
// Flash-style causal attention.  grid = (S/128, B*H), 8 waves, 16 queries/wave.
// scores scaled by 1/sqrt(E) (=768) exactly as reference.
// V consumed from transposed copy vt[(b*H+h)*64+d][s] -> vector B fragments.
// ---------------------------------------------------------------------------
__global__ __launch_bounds__(256) void attn_kernel(
    const _Float16* __restrict__ q, const _Float16* __restrict__ k,
    const _Float16* __restrict__ vt, _Float16* __restrict__ ctx)
{
  __shared__ _Float16 pbuf[8][16][32];     // per-wave P scratch (C->A relayout)

  const int t    = threadIdx.x;
  const int lane = t & 31, wave = t >> 5;
  const int nl   = lane & 15, hi = lane >> 4;
  const int bh   = blockIdx.y;
  const int b    = bh / H_;
  const int h    = bh % H_;
  const int qpos = blockIdx.x * 128 + wave * 16;   // position in sequence
  const int qtok = b * S_ + qpos;
  const int hcol = h * D_;
  const float scale  = 0.03608439182435161f;       // 1/sqrt(768)
  const float NEGBIG = -3.0e38f;

  // ---- Q fragments (d 0..31 and 32..63) ----
  HF16 qa[2];
#pragma unroll
  for (int f = 0; f < 2; ++f) {
    const _Float16* base = q + (size_t)(qtok + nl) * E_ + hcol + f * 32;
    qa[f].h[0] = *(const v8h*)(base + hi * 8);
    qa[f].h[1] = *(const v8h*)(base + 16 + hi * 8);
  }

  const v8f zero8 = {0.f, 0.f, 0.f, 0.f, 0.f, 0.f, 0.f, 0.f};
  v8f o[4];
#pragma unroll
  for (int j = 0; j < 4; ++j) o[j] = zero8;
  float mrow[8], lrow[8];
#pragma unroll
  for (int r = 0; r < 8; ++r) { mrow[r] = NEGBIG; lrow[r] = 0.f; }

  const int nchunk = ((qpos + 15) >> 5) + 1;       // keys up to qpos+15
  for (int ch = 0; ch < nchunk; ++ch) {
    const int kv0 = ch * 32;

    // ---- S = Q K^T  (two 16-key tiles, K-dim = D in two steps of 32) ----
    v8f sc[2] = {zero8, zero8};
#pragma unroll
    for (int tt = 0; tt < 2; ++tt) {
#pragma unroll
      for (int s2 = 0; s2 < 2; ++s2) {
        HF16 kf;
        const _Float16* base =
            k + (size_t)(b * S_ + kv0 + tt * 16 + nl) * E_ + hcol + s2 * 32;
        kf.h[0] = *(const v8h*)(base + hi * 8);
        kf.h[1] = *(const v8h*)(base + 16 + hi * 8);
        sc[tt] = WMMA_F16(qa[s2].v, kf.v, sc[tt]);
      }
    }

    // ---- online softmax update ----
    float pv0[8], pv1[8], alpha[8];
#pragma unroll
    for (int r = 0; r < 8; ++r) {
      const int qrow = qpos + r + 8 * hi;
      float s0 = sc[0][r] * scale;
      float s1 = sc[1][r] * scale;
      if (kv0 + nl > qrow)      s0 = NEGBIG;
      if (kv0 + 16 + nl > qrow) s1 = NEGBIG;
      float mx = fmaxf(s0, s1);
#pragma unroll
      for (int m = 1; m < 16; m <<= 1) mx = fmaxf(mx, __shfl_xor(mx, m, 32));
      const float mnew = fmaxf(mrow[r], mx);
      alpha[r] = __expf(mrow[r] - mnew);
      const float p0 = __expf(s0 - mnew);
      const float p1 = __expf(s1 - mnew);
      pv0[r] = p0; pv1[r] = p1;
      float ps = p0 + p1;
#pragma unroll
      for (int m = 1; m < 16; m <<= 1) ps += __shfl_xor(ps, m, 32);
      lrow[r] = lrow[r] * alpha[r] + ps;
      mrow[r] = mnew;
    }

    // ---- rescale O, write P (C-layout) to LDS, reload in A-layout ----
#pragma unroll
    for (int j = 0; j < 4; ++j)
#pragma unroll
      for (int r = 0; r < 8; ++r) o[j][r] *= alpha[r];
#pragma unroll
    for (int r = 0; r < 8; ++r) {
      pbuf[wave][r + 8 * hi][nl]      = (_Float16)pv0[r];
      pbuf[wave][r + 8 * hi][16 + nl] = (_Float16)pv1[r];
    }
    HF16 pa;
    {
      const _Float16* pr = &pbuf[wave][nl][0];
      pa.h[0] = *(const v8h*)(pr + hi * 8);
      pa.h[1] = *(const v8h*)(pr + 16 + hi * 8);
    }

    // ---- O += P V  (four 16-wide d tiles, vector B fragments from vt) ----
#pragma unroll
    for (int j = 0; j < 4; ++j) {
      HF16 vf;
      const _Float16* vp =
          vt + ((size_t)bh * 64 + j * 16 + nl) * S_ + kv0;
      vf.h[0] = *(const v8h*)(vp + hi * 8);
      vf.h[1] = *(const v8h*)(vp + 16 + hi * 8);
      o[j] = WMMA_F16(pa.v, vf.v, o[j]);
    }
  }

  // ---- normalize + store ctx (f16, [token][E] layout) ----
#pragma unroll
  for (int j = 0; j < 4; ++j)
#pragma unroll
    for (int r = 0; r < 8; ++r) {
      const int row = qtok + r + 8 * hi;
      ctx[(size_t)row * E_ + hcol + j * 16 + nl] = (_Float16)(o[j][r] / lrow[r]);
    }
}

// ---------------------------------------------------------------------------
// Host orchestration
// ---------------------------------------------------------------------------
extern "C" void kernel_launch(void* const* d_in, const int* in_sizes, int n_in,
                              void* d_out, int out_size, void* d_ws, size_t ws_size,
                              hipStream_t stream) {
  const int*   tokens  = (const int*)  d_in[0];
  const float* tok_emb = (const float*)d_in[1];
  const float* pos_emb = (const float*)d_in[2];
  const float* ln1_s   = (const float*)d_in[3];
  const float* ln1_b   = (const float*)d_in[4];
  const float* Wq      = (const float*)d_in[5];
  const float* bq      = (const float*)d_in[6];
  const float* Wk      = (const float*)d_in[7];
  const float* bk      = (const float*)d_in[8];
  const float* Wv      = (const float*)d_in[9];
  const float* bv      = (const float*)d_in[10];
  const float* Wo      = (const float*)d_in[11];
  const float* bo      = (const float*)d_in[12];
  const float* ln2_s   = (const float*)d_in[13];
  const float* ln2_b   = (const float*)d_in[14];
  const float* W1      = (const float*)d_in[15];
  const float* b1      = (const float*)d_in[16];
  const float* W2      = (const float*)d_in[17];
  const float* b2      = (const float*)d_in[18];
  const float* lnf_s   = (const float*)d_in[19];
  const float* lnf_b   = (const float*)d_in[20];
  const float* Whead   = (const float*)d_in[21];
  float* out = (float*)d_out;

  char* wp = (char*)d_ws;
  auto alloc = [&](size_t bytes) {
    void* p = (void*)wp;
    wp += (bytes + 255) & ~(size_t)255;
    return p;
  };
  float*     x    = (float*)    alloc((size_t)NTOK * E_ * 4);
  _Float16*  y16  = (_Float16*) alloc((size_t)NTOK * E_ * 2);
  _Float16*  q16  = (_Float16*) alloc((size_t)NTOK * E_ * 2);
  _Float16*  k16  = (_Float16*) alloc((size_t)NTOK * E_ * 2);
  _Float16*  v16  = (_Float16*) alloc((size_t)NTOK * E_ * 2);
  _Float16*  c16  = (_Float16*) alloc((size_t)NTOK * E_ * 2);
  _Float16*  vt16 = (_Float16*) alloc((size_t)NTOK * E_ * 2);   // transposed V
  _Float16*  h16  = (_Float16*) alloc((size_t)NTOK * 4 * E_ * 2);

  const dim3 blk(256);
  const dim3 g768(E_ / BN, NTOK / BM);              // 6 x 16
  const dim3 g3072(4 * E_ / BN, NTOK / BM);         // 24 x 16
  const dim3 ghead((V_ + BN - 1) / BN, NTOK / BM);  // 393 x 16
  const dim3 gattn(S_ / 128, 2 * H_);               // 8 x 24
  const dim3 gvt(S_ / 64, H_, 2);                   // 16 x 12 x 2

  embed_kernel<<<NTOK, blk, 0, stream>>>(tokens, tok_emb, pos_emb, x);

  for (int l = 0; l < L_; ++l) {
    const size_t wEE = (size_t)l * E_ * E_;
    const size_t wE4 = (size_t)l * E_ * 4 * E_;
    ln_kernel<<<NTOK, blk, 0, stream>>>(x, ln1_s + l * E_, ln1_b + l * E_, y16);
    gemm_kernel<false, false, true, true><<<g768, blk, 0, stream>>>(
        y16, Wq + wEE, bq + l * E_, nullptr, nullptr, q16, NTOK, E_, E_);
    gemm_kernel<false, false, true, true><<<g768, blk, 0, stream>>>(
        y16, Wk + wEE, bk + l * E_, nullptr, nullptr, k16, NTOK, E_, E_);
    gemm_kernel<false, false, true, true><<<g768, blk, 0, stream>>>(
        y16, Wv + wEE, bv + l * E_, nullptr, nullptr, v16, NTOK, E_, E_);
    vtrans_kernel<<<gvt, blk, 0, stream>>>(v16, vt16);
    attn_kernel<<<gattn, blk, 0, stream>>>(q16, k16, vt16, c16);
    gemm_kernel<false, true, false, true><<<g768, blk, 0, stream>>>(
        c16, Wo + wEE, bo + l * E_, x, x, nullptr, NTOK, E_, E_);
    ln_kernel<<<NTOK, blk, 0, stream>>>(x, ln2_s + l * E_, ln2_b + l * E_, y16);
    gemm_kernel<true, false, true, true><<<g3072, blk, 0, stream>>>(
        y16, W1 + wE4, b1 + l * 4 * E_, nullptr, nullptr, h16, NTOK, 4 * E_, E_);
    gemm_kernel<false, true, false, true><<<g768, blk, 0, stream>>>(
        h16, W2 + wE4, b2 + l * E_, x, x, nullptr, NTOK, E_, 4 * E_);
  }

  ln_kernel<<<NTOK, blk, 0, stream>>>(x, lnf_s, lnf_b, y16);
  gemm_kernel<false, false, false, false><<<ghead, blk, 0, stream>>>(
      y16, Whead, nullptr, nullptr, out, nullptr, NTOK, V_, E_);
}
